// Attention_90769838833757
// MI455X (gfx1250) — compile-verified, matrix operand order checked by reading, not run on hardware
//
#include <hip/hip_runtime.h>
#include <hip/hip_bf16.h>

// B=2, S=2048, D=1024, H=16, HD=64
#define BATCH 2
#define SEQ   2048
#define DIM   1024
#define HEADS 16
#define HDIM  64
#define EPS   1e-6f

typedef __attribute__((ext_vector_type(16))) __bf16 v16bf;
typedef __attribute__((ext_vector_type(2)))  __bf16 v2bf;
typedef __attribute__((ext_vector_type(8)))  float  v8f;

__device__ __forceinline__ unsigned short f2bf(float x) {
  return __builtin_bit_cast(unsigned short, (__bf16)x);   // native cvt, RNE
}
__device__ __forceinline__ unsigned int pk2bf(float lo, float hi) {
#if __has_builtin(__builtin_amdgcn_cvt_pk_bf16_f32)
  return __builtin_bit_cast(unsigned int,
                            __builtin_amdgcn_cvt_pk_bf16_f32(lo, hi));
#else
  return (unsigned)f2bf(lo) | ((unsigned)f2bf(hi) << 16);
#endif
}

// 16x32 bf16 fragment in the CDNA5 16-bit A-matrix layout (§7.12.2).
// Per lane the 8 K-pairs live at ushort offsets {0,2,4,6}+8*hi and
// {16,18,20,22}+8*hi -> two aligned b128 loads.
// `stride` (ushorts) must be a multiple of 8 and base 16B-aligned.
__device__ __forceinline__ v16bf frag16(const unsigned short* p0, int stride, int lane) {
  const int r  = lane & 15;
  const int hi = (lane >> 4) & 1;
  const unsigned short* p = p0 + r * stride + (hi << 3);
  union { uint4 q[2]; v16bf v; } f;
  f.q[0] = *(const uint4*)(p);
  f.q[1] = *(const uint4*)(p + 16);
  return f.v;
}

#define WMMA_BF16(a, b, c) __builtin_amdgcn_wmma_f32_16x16x32_bf16( \
    false, (a), false, (b), (short)0, (c), false, false)

// ---------------------------------------------------------------------------
// Kernel 1: qkv = x @ Wqkv + b, fused RMSNorm + RoPE epilogue, bf16 outputs.
// 128x64 tile, BK=32, 8 waves x 4 WMMA accums, double-buffered LDS pipeline.
// ---------------------------------------------------------------------------
__global__ __launch_bounds__(256) void qkv_norm_rope_kernel(
    const float* __restrict__ x, const float* __restrict__ W,
    const float* __restrict__ bias, const float* __restrict__ qg,
    const float* __restrict__ kg, const float* __restrict__ cosT,
    const float* __restrict__ sinT, unsigned short* __restrict__ wq,
    unsigned short* __restrict__ wk, unsigned short* __restrict__ wv) {
  __shared__ __align__(16) unsigned char smem[33280];  // max(A2+B2, C) bytes

  const int tid  = threadIdx.x;
  const int wave = tid >> 5;
  const int lane = tid & 31;
  const int n0   = blockIdx.x * 64;
  const int m0   = blockIdx.y * 128;

  // LDS addresses computed per-use (pointer arrays over LDS would emit an
  // addrspacecast static initializer that ld.lld rejects).
  auto bufA = [&](int b) -> unsigned short* {
    return (unsigned short*)smem + b * 5120;           // 128*40 each
  };
  auto bufB = [&](int b) -> unsigned short* {
    return (unsigned short*)smem + 10240 + b * 2560;   // 64*40 each
  };

  auto stageA = [&](int buf, int k0) {                 // 128x32 fp32 -> bf16
    unsigned short* A = bufA(buf);
    const float4* xs = (const float4*)(x + (size_t)m0 * DIM + k0);
    float4 v[4];
#pragma unroll
    for (int i = 0; i < 4; ++i) {                      // batch all loads
      int c = i * 256 + tid;
      v[i] = xs[(size_t)(c >> 3) * (DIM / 4) + (c & 7)];
    }
#pragma unroll
    for (int i = 0; i < 4; ++i) {                      // then convert+store
      int c = i * 256 + tid;
      int r = c >> 3, kq = c & 7;
      uint2 u;
      u.x = pk2bf(v[i].x, v[i].y);
      u.y = pk2bf(v[i].z, v[i].w);
      *(uint2*)(A + r * 40 + kq * 4) = u;
    }
  };
  auto stageB = [&](int buf, int k0) {                 // 32x64 -> [n][k] bf16
    unsigned short* Bt = bufB(buf);
    const float4* ws = (const float4*)(W + (size_t)k0 * (3 * DIM) + n0);
    float4 v[2];
#pragma unroll
    for (int i = 0; i < 2; ++i) {
      int c = i * 256 + tid;
      v[i] = ws[(size_t)(c >> 4) * (3 * DIM / 4) + (c & 15)];
    }
#pragma unroll
    for (int i = 0; i < 2; ++i) {
      int c = i * 256 + tid;
      int kk = c >> 4, nq = (c & 15) * 4;
      Bt[(nq + 0) * 40 + kk] = f2bf(v[i].x);
      Bt[(nq + 1) * 40 + kk] = f2bf(v[i].y);
      Bt[(nq + 2) * 40 + kk] = f2bf(v[i].z);
      Bt[(nq + 3) * 40 + kk] = f2bf(v[i].w);
    }
  };

  v8f acc[4] = {};
  stageA(0, 0);
  stageB(0, 0);
  for (int kt = 0; kt < 32; ++kt) {
    __syncthreads();
    const unsigned short* Ac = bufA(kt & 1);
    const unsigned short* Bc = bufB(kt & 1);
    // Preload fragments; WMMAs issue back-to-back behind one dscnt wait.
    v16bf a  = frag16(Ac + wave * 16 * 40, 40, lane);
    v16bf b0 = frag16(Bc +  0 * 640, 40, lane);
    v16bf b1 = frag16(Bc +  1 * 640, 40, lane);
    v16bf b2 = frag16(Bc +  2 * 640, 40, lane);
    v16bf b3 = frag16(Bc +  3 * 640, 40, lane);
    acc[0] = WMMA_BF16(a, b0, acc[0]);
    acc[1] = WMMA_BF16(a, b1, acc[1]);
    acc[2] = WMMA_BF16(a, b2, acc[2]);
    acc[3] = WMMA_BF16(a, b3, acc[3]);
    if (kt + 1 < 32) {                 // stage next tile while WMMAs drain
      stageA((kt + 1) & 1, (kt + 1) * 32);
      stageB((kt + 1) & 1, (kt + 1) * 32);
    }
  }
  __syncthreads();                      // staging buffers dead; reuse as ldsC

  float* ldsC = (float*)smem;           // 128 x 65 fp32
  const int hi = (lane >> 4) & 1;
#pragma unroll
  for (int t = 0; t < 4; ++t) {
    int n = (lane & 15) + t * 16;
    float bv = bias[n0 + n];
#pragma unroll
    for (int r = 0; r < 8; ++r) {
      int m = wave * 16 + r + hi * 8;
      ldsC[m * 65 + n] = acc[t][r] + bv;
    }
  }
  __syncthreads();

  const int sec = n0 >> 10;            // 0=q, 1=k, 2=v
  const int h   = (n0 & 1023) >> 6;
  if (tid < 128) {
    const int m  = tid;
    const int gm = m0 + m;
    const int b  = gm >> 11;
    const int s  = gm & (SEQ - 1);
    const float* row = ldsC + m * 65;
    const size_t dstoff = (((size_t)(b * HEADS + h) * SEQ + s) << 6);
    if (sec == 2) {
      unsigned short* dst = wv + dstoff;
#pragma unroll 8
      for (int d = 0; d < HDIM; ++d) dst[d] = f2bf(row[d]);
    } else {
      const float* g = (sec == 0) ? qg : kg;
      unsigned short* dst = ((sec == 0) ? wq : wk) + dstoff;
      float ss = 0.f;
#pragma unroll 8
      for (int d = 0; d < HDIM; ++d) { float v = row[d]; ss += v * v; }
      const float rms = rsqrtf(ss * (1.0f / HDIM) + EPS);
      const float* cs = cosT + s * HDIM;
      const float* sn = sinT + s * HDIM;
#pragma unroll 8
      for (int d = 0; d < HDIM; ++d) {
        int pd = (d < 32) ? d + 32 : d - 32;
        float v = row[d]  * rms * g[d];
        float o = row[pd] * rms * g[pd];
        float res = (d < 32) ? (v * cs[d] - o * sn[d])
                             : (v * cs[d] + o * sn[d]);
        dst[d] = f2bf(res);
      }
    }
  }
}

// ---------------------------------------------------------------------------
// Kernel 2: flash attention per (b,h). 128-query tile (16 rows/wave),
// KV blocks of 32 keys, double-buffered K/V staging.
// ---------------------------------------------------------------------------
__global__ __launch_bounds__(256) void flash_attn_kernel(
    const unsigned short* __restrict__ wq, const unsigned short* __restrict__ wk,
    const unsigned short* __restrict__ wv, unsigned short* __restrict__ wo) {
  __shared__ __align__(16) unsigned short ldsK[2][32 * 72];  // [key][dim]
  __shared__ __align__(16) unsigned short ldsV[2][64 * 40];  // [dim][key]
  __shared__ __align__(16) unsigned short ldsP[8 * 16 * 40]; // per-wave P

  const int tid  = threadIdx.x;
  const int wave = tid >> 5;
  const int lane = tid & 31;
  const int bh   = blockIdx.y;
  const int b    = bh >> 4, h = bh & 15;
  const int q0   = blockIdx.x * 128;

  const size_t base = (size_t)bh * SEQ * HDIM;
  const unsigned short* qp = wq + base;
  const unsigned short* kp = wk + base;
  const unsigned short* vp = wv + base;

  // Q fragments straight from global: A-layout rows == lanes, two b128 each.
  const int qr = q0 + wave * 16;
  const v16bf qa0 = frag16(qp + (size_t)qr * HDIM,      HDIM, lane);
  const v16bf qa1 = frag16(qp + (size_t)qr * HDIM + 32, HDIM, lane);

  auto stageKV = [&](int buf, int key0) {
    const int key = tid >> 3;
    const int sub = tid & 7;
    // Batch both global loads, then store.
    uint4 kd = ((const uint4*)(kp + (size_t)key0 * HDIM))[key * 8 + sub];
    union { uint4 q; unsigned short s[8]; } u;
    u.q = *(const uint4*)(vp + (size_t)(key0 + key) * HDIM + sub * 8);
    ((uint4*)ldsK[buf])[key * 9 + sub] = kd;           // stride 9*uint4 = 72
    unsigned short* Vt = ldsV[buf];
    const int dq = sub * 8;
#pragma unroll
    for (int j = 0; j < 8; ++j) Vt[(dq + j) * 40 + key] = u.s[j];
  };

  v8f o[4] = {};
  float mrow[8], lrow[8];
#pragma unroll
  for (int r = 0; r < 8; ++r) { mrow[r] = -3.0e38f; lrow[r] = 0.f; }
  unsigned short* Pw = ldsP + wave * 16 * 40;

  stageKV(0, 0);
  for (int kb = 0; kb < SEQ / 32; ++kb) {
    __syncthreads();
    const int cur = kb & 1;
    const unsigned short* Kt = ldsK[cur];
    const unsigned short* Vt = ldsV[cur];

    // Preload all 4 K fragments, then 4 back-to-back WMMAs.
    v16bf kb00 = frag16(Kt,                72, lane);
    v16bf kb01 = frag16(Kt + 32,           72, lane);
    v16bf kb10 = frag16(Kt + 16 * 72,      72, lane);
    v16bf kb11 = frag16(Kt + 16 * 72 + 32, 72, lane);
    v8f sacc[2] = {};
    sacc[0] = WMMA_BF16(qa0, kb00, sacc[0]);
    sacc[0] = WMMA_BF16(qa1, kb01, sacc[0]);
    sacc[1] = WMMA_BF16(qa0, kb10, sacc[1]);
    sacc[1] = WMMA_BF16(qa1, kb11, sacc[1]);

    // Online softmax; C-layout row m = r + 8*(lane>=16), cols across 16 lanes
#pragma unroll
    for (int r = 0; r < 8; ++r) {
      float v0 = sacc[0][r] * 0.125f;   // 1/sqrt(HD)
      float v1 = sacc[1][r] * 0.125f;
      float tmax = fmaxf(v0, v1);
#pragma unroll
      for (int mask = 1; mask < 16; mask <<= 1)
        tmax = fmaxf(tmax, __shfl_xor(tmax, mask, 32));
      float mnew  = fmaxf(mrow[r], tmax);
      float alpha = __expf(mrow[r] - mnew);
      mrow[r] = mnew;
      float p0 = __expf(v0 - mnew);
      float p1 = __expf(v1 - mnew);
      float ts = p0 + p1;
#pragma unroll
      for (int mask = 1; mask < 16; mask <<= 1)
        ts += __shfl_xor(ts, mask, 32);
      lrow[r] = lrow[r] * alpha + ts;
#pragma unroll
      for (int t = 0; t < 4; ++t) o[t][r] *= alpha;
      int m = r + ((lane >> 4) & 1) * 8;
      int c = lane & 15;
      Pw[m * 40 + c]      = f2bf(p0);
      Pw[m * 40 + c + 16] = f2bf(p1);
    }
    asm volatile("s_wait_dscnt 0" ::: "memory");  // wave-local P relayout

    v16bf pa  = frag16(Pw, 40, lane);
    v16bf vb0 = frag16(Vt + 0 * 640, 40, lane);
    v16bf vb1 = frag16(Vt + 1 * 640, 40, lane);
    v16bf vb2 = frag16(Vt + 2 * 640, 40, lane);
    v16bf vb3 = frag16(Vt + 3 * 640, 40, lane);
    o[0] = WMMA_BF16(pa, vb0, o[0]);
    o[1] = WMMA_BF16(pa, vb1, o[1]);
    o[2] = WMMA_BF16(pa, vb2, o[2]);
    o[3] = WMMA_BF16(pa, vb3, o[3]);
    if (kb + 1 < SEQ / 32) stageKV((kb + 1) & 1, (kb + 1) * 32);
  }

  // Normalize and write bf16 to [B, S, H*HD] for the projection GEMM.
  const int hi = (lane >> 4) & 1;
#pragma unroll
  for (int r = 0; r < 8; ++r) {
    int m  = r + hi * 8;
    int gq = q0 + wave * 16 + m;
    float inv = 1.0f / lrow[r];
    size_t off = ((size_t)(b * SEQ + gq) * DIM) + h * HDIM + (lane & 15);
#pragma unroll
    for (int t = 0; t < 4; ++t) wo[off + t * 16] = f2bf(o[t][r] * inv);
  }
}

// ---------------------------------------------------------------------------
// Kernel 3: out = attn @ Wproj + bproj (fp32 output), double-buffered.
// ---------------------------------------------------------------------------
__global__ __launch_bounds__(256) void proj_kernel(
    const unsigned short* __restrict__ A, const float* __restrict__ W,
    const float* __restrict__ bias, float* __restrict__ out) {
  __shared__ __align__(16) unsigned short ldsA[2][128 * 40];
  __shared__ __align__(16) unsigned short ldsB[2][64 * 40];
  const int tid  = threadIdx.x;
  const int wave = tid >> 5;
  const int lane = tid & 31;
  const int n0   = blockIdx.x * 64;
  const int m0   = blockIdx.y * 128;

  auto stageA = [&](int buf, int k0) {   // A already bf16: uint4 copies
    const uint4* src = (const uint4*)A;
    uint4 v[2];
#pragma unroll
    for (int i = 0; i < 2; ++i) {
      int c = i * 256 + tid;
      v[i] = src[(size_t)(m0 + (c >> 2)) * (DIM / 8) + (k0 >> 3) + (c & 3)];
    }
    uint4* dst = (uint4*)ldsA[buf];
#pragma unroll
    for (int i = 0; i < 2; ++i) {
      int c = i * 256 + tid;
      dst[(c >> 2) * 5 + (c & 3)] = v[i];
    }
  };
  auto stageB = [&](int buf, int k0) {   // W fp32 -> bf16, transposed
    const float4* ws = (const float4*)(W + (size_t)k0 * DIM + n0);
    float4 v[2];
#pragma unroll
    for (int i = 0; i < 2; ++i) {
      int c = i * 256 + tid;
      v[i] = ws[(size_t)(c >> 4) * (DIM / 4) + (c & 15)];
    }
    unsigned short* Bt = ldsB[buf];
#pragma unroll
    for (int i = 0; i < 2; ++i) {
      int c = i * 256 + tid;
      int kk = c >> 4, nq = (c & 15) * 4;
      Bt[(nq + 0) * 40 + kk] = f2bf(v[i].x);
      Bt[(nq + 1) * 40 + kk] = f2bf(v[i].y);
      Bt[(nq + 2) * 40 + kk] = f2bf(v[i].z);
      Bt[(nq + 3) * 40 + kk] = f2bf(v[i].w);
    }
  };

  v8f acc[4] = {};
  stageA(0, 0);
  stageB(0, 0);
  for (int kt = 0; kt < 32; ++kt) {
    __syncthreads();
    const unsigned short* Ac = ldsA[kt & 1];
    const unsigned short* Bc = ldsB[kt & 1];
    v16bf a  = frag16(Ac + wave * 16 * 40, 40, lane);
    v16bf b0 = frag16(Bc + 0 * 640, 40, lane);
    v16bf b1 = frag16(Bc + 1 * 640, 40, lane);
    v16bf b2 = frag16(Bc + 2 * 640, 40, lane);
    v16bf b3 = frag16(Bc + 3 * 640, 40, lane);
    acc[0] = WMMA_BF16(a, b0, acc[0]);
    acc[1] = WMMA_BF16(a, b1, acc[1]);
    acc[2] = WMMA_BF16(a, b2, acc[2]);
    acc[3] = WMMA_BF16(a, b3, acc[3]);
    if (kt + 1 < 32) {
      stageA((kt + 1) & 1, (kt + 1) * 32);
      stageB((kt + 1) & 1, (kt + 1) * 32);
    }
  }

  const int hi = (lane >> 4) & 1;
#pragma unroll
  for (int t = 0; t < 4; ++t) {
    int n = (lane & 15) + t * 16;
    float bv = bias[n0 + n];
#pragma unroll
    for (int r = 0; r < 8; ++r) {
      int m = wave * 16 + r + hi * 8;
      out[(size_t)(m0 + m) * DIM + n0 + n] = acc[t][r] + bv;
    }
  }
}

// ---------------------------------------------------------------------------
extern "C" void kernel_launch(void* const* d_in, const int* in_sizes, int n_in,
                              void* d_out, int out_size, void* d_ws, size_t ws_size,
                              hipStream_t stream) {
  (void)in_sizes; (void)n_in; (void)out_size; (void)ws_size;
  const float* x     = (const float*)d_in[0];
  const float* Wqkv  = (const float*)d_in[1];
  const float* bqkv  = (const float*)d_in[2];
  const float* qg    = (const float*)d_in[3];
  const float* kg    = (const float*)d_in[4];
  const float* Wproj = (const float*)d_in[5];
  const float* bproj = (const float*)d_in[6];
  const float* cosT  = (const float*)d_in[7];
  const float* sinT  = (const float*)d_in[8];

  const size_t qkvElems = (size_t)BATCH * HEADS * SEQ * HDIM;
  unsigned short* wq = (unsigned short*)d_ws;
  unsigned short* wk = wq + qkvElems;
  unsigned short* wv = wk + qkvElems;
  unsigned short* wo = wv + qkvElems;     // [B,S,D] bf16
  float* out = (float*)d_out;

  qkv_norm_rope_kernel<<<dim3(48, 32), 256, 0, stream>>>(
      x, Wqkv, bqkv, qg, kg, cosT, sinT, wq, wk, wv);
  flash_attn_kernel<<<dim3(16, 32), 256, 0, stream>>>(wq, wk, wv, wo);
  proj_kernel<<<dim3(16, 32), 256, 0, stream>>>(wo, Wproj, bproj, out);
}
